// Attention_22591527977475
// MI455X (gfx1250) — matrix operand order repaired and checked by reading, not compiled, not run on hardware
//
#include <hip/hip_runtime.h>
#include <stdint.h>

// Problem constants (match reference)
#define B_SZ 256
#define L_SZ 2048
#define H_SZ 240
#define KPAD 256          // K padded to multiple of 32 for wmma 16x16x32
#define PITCH 264         // LDS row pitch (elems): 132 dwords -> +4 banks/row, conflict-free frags
#define THREADS 512
#define NWAVES (THREADS/32)
#define ROWS 32           // L rows per tile (2 x 16-row WMMA sub-tiles)
#define NT 15             // H / 16 output column tiles

typedef __attribute__((ext_vector_type(16))) __bf16 v16bf;
typedef __attribute__((ext_vector_type(8)))  float  v8f;

union Frag { unsigned u[8]; v16bf v; };

__device__ __forceinline__ unsigned short f2bf(float f) {
  // round-to-nearest-even fp32 -> bf16
  unsigned u = __builtin_bit_cast(unsigned, f);
  unsigned r = 0x7FFFu + ((u >> 16) & 1u);
  return (unsigned short)((u + r) >> 16);
}

__global__ __launch_bounds__(THREADS)
void attn_tanh_softmax_kernel(const float* __restrict__ h,    // [B,L,H]
                              const float* __restrict__ u_s,  // [B,H]
                              const float* __restrict__ W,    // [H,H]
                              const float* __restrict__ bias, // [H]
                              const int*   __restrict__ length, // [B]
                              float* __restrict__ out)        // [B,H]
{
  extern __shared__ char smem[];
  unsigned short* Wl = (unsigned short*)smem;                // [H_SZ][PITCH] bf16 (W[o][k])
  unsigned short* Hl = Wl + H_SZ * PITCH;                    // [ROWS][PITCH] bf16 (h tile)
  float* sc  = (float*)(Hl + ROWS * PITCH);                  // [L_SZ] scores -> alpha
  float* usl = sc + L_SZ;                                    // [256] u_s (this batch)
  float* bl  = usl + 256;                                    // [256] bias
  float* red = bl + 256;                                     // [32] reduction scratch

  const int tid   = threadIdx.x;
  const int lane  = tid & 31;
  const int wid   = tid >> 5;
  const int batch = blockIdx.x;
  const int len   = length[batch];
  const float* hb = h + (size_t)batch * L_SZ * H_SZ;

  // ---- stage W (bf16, K padded with zeros), u_s, bias; zero scores ----
  for (int idx = tid; idx < H_SZ * PITCH; idx += THREADS) {
    int o = idx / PITCH, k = idx - o * PITCH;
    float w = (k < H_SZ) ? W[o * H_SZ + k] : 0.0f;
    Wl[idx] = f2bf(w);
  }
  if (tid < H_SZ) { usl[tid] = u_s[batch * H_SZ + tid]; bl[tid] = bias[tid]; }
  for (int i = tid; i < L_SZ; i += THREADS) sc[i] = 0.0f;
  __syncthreads();

  const int nTiles = (len + ROWS - 1) / ROWS;
  const int half   = lane >> 4;    // half-wave select
  const int mloc   = lane & 15;    // A-row within sub-tile / D column n
  const unsigned* Hl32 = (const unsigned*)Hl;
  const unsigned* Wl32 = (const unsigned*)Wl;

  // ================= Pass 1: scores[l] = u_s . tanh(h W^T + b) =================
  for (int tile = 0; tile < nTiles; ++tile) {
    const int l0 = tile * ROWS;
    // stage h tile -> LDS bf16 (K padded)
    for (int idx = tid; idx < ROWS * PITCH; idx += THREADS) {
      int r = idx / PITCH, k = idx - r * PITCH;
      float x = (k < H_SZ) ? hb[(size_t)(l0 + r) * H_SZ + k] : 0.0f;
      Hl[idx] = f2bf(x);
    }
    if (tile + 1 < nTiles)  // hint next tile into cache (global_prefetch_b8)
      __builtin_prefetch(hb + (size_t)(l0 + ROWS) * H_SZ + tid, 0, 0);
    __syncthreads();

    // 2 (M sub-tiles) x 15 (N tiles) wave tasks over 16 waves
    for (int t = wid; t < 2 * NT; t += NWAVES) {
      const int mt   = t / NT;        // 0..1
      const int nt   = t - mt * NT;   // 0..14
      const int arow = mt * 16 + mloc;
      const int n0   = nt * 16;
      v8f acc = {};
      #pragma unroll
      for (int kb = 0; kb < KPAD / 32; ++kb) {
        const int kk = kb * 32;
        Frag fa, fb;
        #pragma unroll
        for (int j = 0; j < 8; ++j) {
          // A 16x32 bf16: lane halves split K {0-7,16-23} vs {8-15,24-31}
          int ka = ((j < 4) ? 2 * j : 16 + 2 * (j - 4)) + (half ? 8 : 0);
          fa.u[j] = Hl32[(arow * PITCH + kk + ka) >> 1];
          // B 32x16 bf16: lane halves split K {0-7,16-23} vs {8-15,24-31} (same as A)
          int kbcol = ((j < 4) ? 2 * j : 16 + 2 * (j - 4)) + (half ? 8 : 0);
          fb.u[j] = Wl32[((n0 + mloc) * PITCH + kk + kbcol) >> 1];
        }
        acc = __builtin_amdgcn_wmma_f32_16x16x32_bf16(
                  false, fa.v, false, fb.v, (short)0, acc, false, false);
      }
      // D layout: VGPR r -> M = r + 8*half, N = lane&15
      #pragma unroll
      for (int r = 0; r < 8; ++r) {
        const int md = r + 8 * half;
        const int o  = n0 + mloc;
        float val = tanhf(acc[r] + bl[o]) * usl[o];
        val += __shfl_xor(val, 1, 32);
        val += __shfl_xor(val, 2, 32);
        val += __shfl_xor(val, 4, 32);
        val += __shfl_xor(val, 8, 32);
        if (mloc == 0) atomicAdd(&sc[l0 + mt * 16 + md], val);  // ds_add_f32
      }
    }
    __syncthreads();
  }

  // ================= mask + softmax over L (in LDS) =================
  for (int i = tid; i < L_SZ; i += THREADS)
    if (i >= len) sc[i] = -1e30f;
  __syncthreads();

  float mymax = -1e30f;
  for (int i = tid; i < L_SZ; i += THREADS) mymax = fmaxf(mymax, sc[i]);
  for (int off = 16; off; off >>= 1) mymax = fmaxf(mymax, __shfl_xor(mymax, off, 32));
  if (lane == 0) red[wid] = mymax;
  __syncthreads();
  if (tid == 0) {
    float m = red[0];
    for (int w = 1; w < NWAVES; ++w) m = fmaxf(m, red[w]);
    red[NWAVES] = m;
  }
  __syncthreads();
  const float gmax = red[NWAVES];

  float mysum = 0.0f;
  for (int i = tid; i < L_SZ; i += THREADS) {
    float e = __expf(sc[i] - gmax);
    sc[i] = e;
    mysum += e;
  }
  for (int off = 16; off; off >>= 1) mysum += __shfl_xor(mysum, off, 32);
  if (lane == 0) red[wid] = mysum;
  __syncthreads();
  if (tid == 0) {
    float s = 0.0f;
    for (int w = 0; w < NWAVES; ++w) s += red[w];
    red[NWAVES] = s;
  }
  __syncthreads();
  const float rinv = 1.0f / red[NWAVES];
  for (int i = tid; i < L_SZ; i += THREADS) sc[i] *= rinv;
  __syncthreads();

  // ================= Pass 2: v[o] = sum_l alpha[l] * h[l,o] =================
  const int grp = tid >> 8;       // two 256-thread groups split L
  const int o   = tid & 255;
  float acc = 0.0f;
  if (o < H_SZ) {
    #pragma unroll 4
    for (int l = grp; l < len; l += 2)
      acc += sc[l] * hb[(size_t)l * H_SZ + o];   // coalesced over o
  }
  float* vtmp = (float*)Hl;       // reuse h-tile LDS
  vtmp[tid] = acc;
  __syncthreads();
  if (tid < H_SZ) out[batch * H_SZ + tid] = vtmp[tid] + vtmp[tid + 256];
}

extern "C" void kernel_launch(void* const* d_in, const int* in_sizes, int n_in,
                              void* d_out, int out_size, void* d_ws, size_t ws_size,
                              hipStream_t stream) {
  const float* h    = (const float*)d_in[0];  // short_perference [1,B,L,H]
  const float* us   = (const float*)d_in[1];  // current_perference [1,B,H]
  const float* W    = (const float*)d_in[2];  // [H,H]
  const float* bias = (const float*)d_in[3];  // [H]
  const int*   len  = (const int*)d_in[5];    // [B]
  float* out = (float*)d_out;                 // [B,H]

  const size_t smem = (size_t)(H_SZ * PITCH + ROWS * PITCH) * sizeof(unsigned short)
                    + (size_t)(L_SZ + 256 + 256 + 32) * sizeof(float);

  attn_tanh_softmax_kernel<<<B_SZ, THREADS, smem, stream>>>(h, us, W, bias, len, out);
}
